// OAMoudle_74775380623542
// MI455X (gfx1250) — matrix operand order repaired and checked by reading, not compiled, not run on hardware
//
#include <hip/hip_runtime.h>
#include <hip/hip_bf16.h>
#include <stdint.h>

typedef __attribute__((ext_vector_type(16))) _Float16 v16h;
typedef __attribute__((ext_vector_type(8)))  float    v8f;

#define BB 4
#define CC 128
#define HH 48
#define WWD 48
#define HWP (HH*WWD)      // 2304
#define NGR 2
#define CG  (CC/NGR)      // 64
#define NHD 4
#define DH  (CC/NHD)      // 32
#define BGR (BB*NGR)      // 8
#define BHD (BB*NHD)      // 16
#define NSMP HWP          // 2304
#define LN_EPS 1e-5f
#define CO_TILE 16

// ---------------- register-blocked 1x1 conv (16 output channels / thread)
// hmode: 0 = f32 only; 1 = also/only f16 head-major (B*NH, Npix, Dh), scaled; 2 = also/only f16 flat
__global__ void k_conv1x1(const float* __restrict__ x, const float* __restrict__ w,
                          const float* __restrict__ bias, float* __restrict__ y,
                          _Float16* __restrict__ yh, int Ci, int Co, int Npix,
                          float scl, int hmode) {
  int p   = blockIdx.x * blockDim.x + threadIdx.x;
  int co0 = blockIdx.y * CO_TILE;
  int n   = blockIdx.z;
  if (p >= Npix) return;
  const float* xp = x + ((size_t)n * Ci) * Npix + p;
  float acc[CO_TILE];
  #pragma unroll
  for (int j = 0; j < CO_TILE; ++j) acc[j] = bias[co0 + j];
  for (int ci = 0; ci < Ci; ++ci) {
    float xv = xp[(size_t)ci * Npix];              // one load feeds 16 FMAs
    #pragma unroll
    for (int j = 0; j < CO_TILE; ++j)
      acc[j] = fmaf(w[(size_t)(co0 + j) * Ci + ci], xv, acc[j]);
  }
  #pragma unroll
  for (int j = 0; j < CO_TILE; ++j) {
    int co = co0 + j;
    if (y) y[((size_t)n * Co + co) * Npix + p] = acc[j];
    if (hmode == 1) {
      int hh = co / DH, dd = co % DH;
      yh[(((size_t)(n * NHD + hh)) * Npix + p) * DH + dd] = (_Float16)(acc[j] * scl);
    } else if (hmode == 2) {
      yh[((size_t)n * Co + co) * Npix + p] = (_Float16)acc[j];
    }
  }
}

// ---------------- 1x1 conv, single output channel (sobel gate projection)
__global__ void k_conv1x1_s(const float* __restrict__ x, const float* __restrict__ w,
                            const float* __restrict__ bias, float* __restrict__ y,
                            int Ci, int Npix) {
  int p = blockIdx.x * blockDim.x + threadIdx.x;
  int n = blockIdx.z;
  if (p >= Npix) return;
  const float* xp = x + ((size_t)n * Ci) * Npix + p;
  float acc = bias[0];
  for (int ci = 0; ci < Ci; ++ci)
    acc = fmaf(w[ci], xp[(size_t)ci * Npix], acc);
  y[(size_t)n * Npix + p] = acc;
}

// ---------------- depthwise 3x3, stride 1, zero pad 1 (cross-correlation, as XLA)
__global__ void k_dwconv3(const float* __restrict__ x, const float* __restrict__ w,
                          const float* __restrict__ bias, float* __restrict__ y,
                          int Nn, int Cc) {
  size_t gid = (size_t)blockIdx.x * blockDim.x + threadIdx.x;
  size_t tot = (size_t)Nn * Cc * HWP;
  if (gid >= tot) return;
  int wp = (int)(gid % WWD);
  int h  = (int)((gid / WWD) % HH);
  int c  = (int)((gid / HWP) % Cc);
  int n  = (int)(gid / ((size_t)HWP * Cc));
  const float* xp = x + ((size_t)n * Cc + c) * HWP;
  const float* wk = w + (size_t)c * 9;
  float acc = bias[c];
  #pragma unroll
  for (int dy = -1; dy <= 1; ++dy) {
    int hy = h + dy; if (hy < 0 || hy >= HH) continue;
    #pragma unroll
    for (int dx = -1; dx <= 1; ++dx) {
      int wx = wp + dx; if (wx < 0 || wx >= WWD) continue;
      acc = fmaf(wk[(dy + 1) * 3 + (dx + 1)], xp[hy * WWD + wx], acc);
    }
  }
  y[gid] = acc;
}

// ---------------- channel LayerNorm (over Cg) + exact GELU
__global__ void k_ln_gelu(const float* __restrict__ t, const float* __restrict__ g,
                          const float* __restrict__ b, float* __restrict__ o) {
  size_t gid = (size_t)blockIdx.x * blockDim.x + threadIdx.x;
  if (gid >= (size_t)BGR * HWP) return;
  int p  = (int)(gid % HWP);
  int bg = (int)(gid / HWP);
  const float* tp = t + ((size_t)bg * CG) * HWP + p;
  float mu = 0.f;
  for (int c = 0; c < CG; ++c) mu += tp[(size_t)c * HWP];
  mu *= (1.0f / CG);
  float m2 = 0.f;
  for (int c = 0; c < CG; ++c) { float d = tp[(size_t)c * HWP] - mu; m2 += d * d; }
  float inv = rsqrtf(m2 * (1.0f / CG) + LN_EPS);
  float* op = o + ((size_t)bg * CG) * HWP + p;
  for (int c = 0; c < CG; ++c) {
    float xn = (tp[(size_t)c * HWP] - mu) * inv * g[c] + b[c];
    op[(size_t)c * HWP] = 0.5f * xn * (1.0f + erff(xn * 0.70710678118654752f));
  }
}

// ---------------- offset pw conv (Cg->2, no bias) + tanh * range, output (bg,p,{y,x})
__global__ void k_offset(const float* __restrict__ t2, const float* __restrict__ w,
                         float* __restrict__ off) {
  size_t gid = (size_t)blockIdx.x * blockDim.x + threadIdx.x;
  if (gid >= (size_t)BGR * HWP) return;
  int p  = (int)(gid % HWP);
  int bg = (int)(gid / HWP);
  const float* tp = t2 + ((size_t)bg * CG) * HWP + p;
  float a0 = 0.f, a1 = 0.f;
  for (int c = 0; c < CG; ++c) {
    float v = tp[(size_t)c * HWP];
    a0 = fmaf(w[c], v, a0);
    a1 = fmaf(w[CG + c], v, a1);
  }
  off[gid * 2 + 0] = tanhf(a0) * (1.0f / (HH - 1));
  off[gid * 2 + 1] = tanhf(a1) * (1.0f / (WWD - 1));
}

// ---------------- Sobel gate + reference grid + pos (writes d_out pos & reference too)
__global__ void k_pos(const float* __restrict__ sb, const float* __restrict__ off,
                      float* __restrict__ posw, float* __restrict__ outPos,
                      float* __restrict__ outRef) {
  size_t gid = (size_t)blockIdx.x * blockDim.x + threadIdx.x;
  if (gid >= (size_t)BGR * HWP) return;
  int p  = (int)(gid % HWP);
  int bg = (int)(gid / HWP);
  int h  = p / WWD, wc = p % WWD;
  const float* sp = sb + (size_t)bg * HWP;
  const float SX[9] = {-1.f, 0.f, 1.f, -2.f, 0.f, 2.f, -1.f, 0.f, 1.f};
  const float SY[9] = {-1.f, -2.f, -1.f, 0.f, 0.f, 0.f, 1.f, 2.f, 1.f};
  float gx = 0.f, gy = 0.f;
  #pragma unroll
  for (int dy = -1; dy <= 1; ++dy) {
    int hy = h + dy; if (hy < 0 || hy >= HH) continue;
    #pragma unroll
    for (int dx = -1; dx <= 1; ++dx) {
      int wx = wc + dx; if (wx < 0 || wx >= WWD) continue;
      float v = sp[hy * WWD + wx];
      gx = fmaf(SX[(dy + 1) * 3 + dx + 1], v, gx);
      gy = fmaf(SY[(dy + 1) * 3 + dx + 1], v, gy);
    }
  }
  float mag = sqrtf(gx * gx + gy * gy);
  float bin = mag > 0.5f ? 1.0f : 0.0f;
  float ry = ((h + 0.5f) / (HH - 1.0f)) * 2.0f - 1.0f;
  float rx = ((wc + 0.5f) / (WWD - 1.0f)) * 2.0f - 1.0f;
  float py = off[gid * 2 + 0] + ry * bin;
  float px = off[gid * 2 + 1] + rx * bin;
  posw[gid * 2 + 0] = py;  posw[gid * 2 + 1] = px;
  outPos[gid * 2 + 0] = py; outPos[gid * 2 + 1] = px;
  outRef[gid * 2 + 0] = ry; outRef[gid * 2 + 1] = rx;
}

// ---------------- bilinear grid sample, align_corners=True, zero padding
__global__ void k_gridsample(const float* __restrict__ x, const float* __restrict__ posw,
                             float* __restrict__ xs) {
  size_t gid = (size_t)blockIdx.x * blockDim.x + threadIdx.x;
  if (gid >= (size_t)BGR * CG * NSMP) return;
  int n  = (int)(gid % NSMP);
  int c  = (int)((gid / NSMP) % CG);
  int bg = (int)(gid / ((size_t)NSMP * CG));
  float py = posw[((size_t)bg * NSMP + n) * 2 + 0];
  float px = posw[((size_t)bg * NSMP + n) * 2 + 1];
  float gx = (px + 1.0f) * 0.5f * (WWD - 1);
  float gy = (py + 1.0f) * 0.5f * (HH - 1);
  float x0f = floorf(gx), y0f = floorf(gy);
  int x0 = (int)x0f, y0 = (int)y0f;
  float fx = gx - x0f, fy = gy - y0f;
  const float* img = x + ((size_t)bg * CG + c) * HWP;
  auto samp = [&](int yy, int xx) -> float {
    if (yy < 0 || yy >= HH || xx < 0 || xx >= WWD) return 0.0f;
    return img[yy * WWD + xx];
  };
  float v00 = samp(y0, x0),     v01 = samp(y0, x0 + 1);
  float v10 = samp(y0 + 1, x0), v11 = samp(y0 + 1, x0 + 1);
  xs[gid] = v00 * (1.f - fy) * (1.f - fx) + v01 * (1.f - fy) * fx
          + v10 * fy * (1.f - fx)         + v11 * fy * fx;
}

#define WMMA_F16(A, Bm, Cm) \
  __builtin_amdgcn_wmma_f32_16x16x32_f16(false, (A), false, (Bm), (short)0, (Cm), false, false)

// ---------------- WMMA attention: one wave32 per (head, 32-query block = 2 tiles)
// two-pass softmax with DEFERRED cross-lane reductions (shuffles only outside loops).
// scale is pre-folded into qh. Fuses /Z + lepe into the epilogue.
__global__ __launch_bounds__(32) void k_attn(const _Float16* __restrict__ qh,
                                             const _Float16* __restrict__ kh,
                                             const _Float16* __restrict__ vt,
                                             const float* __restrict__ lepe,
                                             float* __restrict__ outb) {
  __shared__ _Float16 Pl[2 * 16 * 32];
  int lane = threadIdx.x;
  int qb2 = blockIdx.x;                // 0..71  (32 queries each)
  int bh = blockIdx.y;                 // 0..15
  int b = bh / NHD, hhead = bh % NHD;
  const _Float16* Q = qh + (size_t)bh * HWP * DH;
  const _Float16* K = kh + (size_t)bh * NSMP * DH;
  const _Float16* V = vt + (size_t)bh * DH * NSMP;
  int m0 = qb2 * 32;
  int mrow = lane & 15;
  bool lo = lane < 16;
  int ab0 = lo ? 0 : 16, ab1 = lo ? 8 : 24;  // A-operand K-chunk bases (contiguous: k = 16h + s)
  int koff = lo ? 0 : 16;                    // B-operand K base per lane half

  // A operands: two Q tiles 16x32
  v16h aq[2];
  #pragma unroll
  for (int t = 0; t < 2; ++t) {
    const _Float16* qrow = Q + (size_t)(m0 + t * 16 + mrow) * DH;
    #pragma unroll
    for (int j = 0; j < 8; ++j) { aq[t][j] = qrow[ab0 + j]; aq[t][8 + j] = qrow[ab1 + j]; }
  }
  v8f zc;
  #pragma unroll
  for (int r = 0; r < 8; ++r) zc[r] = 0.f;

  // ---- pass 1: per-lane partial row max (no shuffles in the loop)
  float rmax[2][8];
  #pragma unroll
  for (int t = 0; t < 2; ++t)
    #pragma unroll
    for (int r = 0; r < 8; ++r) rmax[t][r] = -3.0e30f;
  for (int kb = 0; kb < NSMP; kb += 32) {
    v16h bk0 = *(const v16h*)(K + (size_t)(kb + mrow) * DH + koff);
    v16h bk1 = *(const v16h*)(K + (size_t)(kb + 16 + mrow) * DH + koff);
    #pragma unroll
    for (int t = 0; t < 2; ++t) {
      v8f s0 = WMMA_F16(aq[t], bk0, zc);
      v8f s1 = WMMA_F16(aq[t], bk1, zc);
      #pragma unroll
      for (int r = 0; r < 8; ++r) rmax[t][r] = fmaxf(rmax[t][r], fmaxf(s0[r], s1[r]));
    }
  }
  // one cross-lane max reduction (within each 16-lane half)
  #pragma unroll
  for (int t = 0; t < 2; ++t)
    #pragma unroll
    for (int r = 0; r < 8; ++r) {
      float m = rmax[t][r];
      #pragma unroll
      for (int off = 1; off <= 8; off <<= 1) m = fmaxf(m, __shfl_xor(m, off, 32));
      rmax[t][r] = m;
    }

  // ---- pass 2: recompute S, exp, per-lane partial Z, accumulate O
  v8f o0[2], o1[2];
  float zacc[2][8];
  #pragma unroll
  for (int t = 0; t < 2; ++t)
    #pragma unroll
    for (int r = 0; r < 8; ++r) { o0[t][r] = 0.f; o1[t][r] = 0.f; zacc[t][r] = 0.f; }
  int mo = lo ? 0 : 8;
  for (int kb = 0; kb < NSMP; kb += 32) {
    v16h bk0 = *(const v16h*)(K + (size_t)(kb + mrow) * DH + koff);
    v16h bk1 = *(const v16h*)(K + (size_t)(kb + 16 + mrow) * DH + koff);
    #pragma unroll
    for (int t = 0; t < 2; ++t) {
      v8f s0 = WMMA_F16(aq[t], bk0, zc);
      v8f s1 = WMMA_F16(aq[t], bk1, zc);
      #pragma unroll
      for (int r = 0; r < 8; ++r) {
        float p0 = __expf(s0[r] - rmax[t][r]);
        float p1 = __expf(s1[r] - rmax[t][r]);
        zacc[t][r] += p0 + p1;                       // per-lane partial Z
        Pl[t * 512 + (r + mo) * 32 + mrow]      = (_Float16)p0;
        Pl[t * 512 + (r + mo) * 32 + 16 + mrow] = (_Float16)p1;
      }
    }
    __syncthreads();
    v16h ap[2];
    #pragma unroll
    for (int t = 0; t < 2; ++t) {
      const _Float16* prow = Pl + t * 512 + mrow * 32;
      #pragma unroll
      for (int j = 0; j < 8; ++j) { ap[t][j] = prow[ab0 + j]; ap[t][8 + j] = prow[ab1 + j]; }
    }
    __syncthreads();
    v16h bv0 = *(const v16h*)(V + (size_t)mrow * NSMP + kb + koff);
    v16h bv1 = *(const v16h*)(V + (size_t)(16 + mrow) * NSMP + kb + koff);
    #pragma unroll
    for (int t = 0; t < 2; ++t) {
      o0[t] = WMMA_F16(ap[t], bv0, o0[t]);
      o1[t] = WMMA_F16(ap[t], bv1, o1[t]);
    }
  }
  // one cross-lane sum reduction for Z
  #pragma unroll
  for (int t = 0; t < 2; ++t)
    #pragma unroll
    for (int r = 0; r < 8; ++r) {
      float zz = zacc[t][r];
      #pragma unroll
      for (int off = 1; off <= 8; off <<= 1) zz += __shfl_xor(zz, off, 32);
      zacc[t][r] = zz;
    }

  // ---- epilogue: /Z + lepe, scatter to (B, C, HW)
  #pragma unroll
  for (int t = 0; t < 2; ++t)
    #pragma unroll
    for (int r = 0; r < 8; ++r) {
      int m = m0 + t * 16 + r + mo;
      float invz = 1.0f / zacc[t][r];
      int c0 = hhead * DH + mrow;
      size_t i0 = ((size_t)b * CC + c0) * HWP + m;
      size_t i1 = ((size_t)b * CC + c0 + 16) * HWP + m;
      outb[i0] = o0[t][r] * invz + lepe[i0];
      outb[i1] = o1[t][r] * invz + lepe[i1];
    }
}

extern "C" void kernel_launch(void* const* d_in, const int* in_sizes, int n_in,
                              void* d_out, int out_size, void* d_ws, size_t ws_size,
                              hipStream_t stream) {
  (void)in_sizes; (void)n_in; (void)out_size; (void)ws_size;
  const float* x     = (const float*)d_in[0];
  const float* qW    = (const float*)d_in[1];
  const float* qb    = (const float*)d_in[2];
  const float* kW    = (const float*)d_in[3];
  const float* kb    = (const float*)d_in[4];
  const float* vW    = (const float*)d_in[5];
  const float* vb    = (const float*)d_in[6];
  const float* oW    = (const float*)d_in[7];
  const float* ob    = (const float*)d_in[8];
  const float* odw   = (const float*)d_in[9];
  const float* odb   = (const float*)d_in[10];
  const float* lng   = (const float*)d_in[11];
  const float* lnb   = (const float*)d_in[12];
  const float* opw   = (const float*)d_in[13];
  const float* sw    = (const float*)d_in[14];
  const float* sbias = (const float*)d_in[15];
  const float* rpw   = (const float*)d_in[16];
  const float* rpb   = (const float*)d_in[17];
  float* out = (float*)d_out;

  char* ws = (char*)d_ws;
  const size_t F4 = (size_t)BB * CC * HWP * sizeof(float);   // 4,718,592 B
  float* q    = (float*)(ws + 0 * F4);
  float* t    = (float*)(ws + 1 * F4);
  float* t2   = (float*)(ws + 2 * F4);
  float* xs   = (float*)(ws + 3 * F4);
  float* lepe = (float*)(ws + 4 * F4);
  float* outb = (float*)(ws + 5 * F4);
  char* p8 = ws + 6 * F4;
  float* off  = (float*)p8; p8 += (size_t)BGR * HWP * 2 * sizeof(float);
  float* sb   = (float*)p8; p8 += (size_t)BGR * HWP * sizeof(float);
  float* posw = (float*)p8; p8 += (size_t)BGR * HWP * 2 * sizeof(float);
  uintptr_t a = ((uintptr_t)p8 + 255) & ~(uintptr_t)255;
  const size_t NE = (size_t)BB * CC * HWP;                   // 1,179,648 elems
  _Float16* qh = (_Float16*)a; a += NE * sizeof(_Float16);
  a = (a + 255) & ~(uintptr_t)255;
  _Float16* kh = (_Float16*)a; a += NE * sizeof(_Float16);
  a = (a + 255) & ~(uintptr_t)255;
  _Float16* vt = (_Float16*)a;

  const float scale = 0.17677669529663687f;  // 32^-0.5
  dim3 blk(256);
  dim3 gconv(HWP / 256, CC / CO_TILE, BB);
  // q projection -> q (f32) + qh (f16 head-major, pre-scaled)
  k_conv1x1<<<gconv, blk, 0, stream>>>(x, qW, qb, q, qh, CC, CC, HWP, scale, 1);
  // offset branch: dwconv -> LN+GELU -> pw+tanh*range
  k_dwconv3<<<(unsigned)(((size_t)BGR * CG * HWP + 255) / 256), blk, 0, stream>>>(q, odw, odb, t, BGR, CG);
  k_ln_gelu<<<(unsigned)(((size_t)BGR * HWP + 255) / 256), blk, 0, stream>>>(t, lng, lnb, t2);
  k_offset<<<(unsigned)(((size_t)BGR * HWP + 255) / 256), blk, 0, stream>>>(t2, opw, off);
  // sobel gate: 1x1 (Cg->1) then sobel mag + pos/reference
  k_conv1x1_s<<<dim3(HWP / 256, 1, BGR), blk, 0, stream>>>(q, sw, sbias, sb, CG, HWP);
  {
    float* outPos = out + NE;
    float* outRef = outPos + (size_t)BGR * HWP * 2;
    k_pos<<<(unsigned)(((size_t)BGR * HWP + 255) / 256), blk, 0, stream>>>(sb, off, posw, outPos, outRef);
  }
  // deformable sampling
  k_gridsample<<<(unsigned)(((size_t)BGR * CG * NSMP + 255) / 256), blk, 0, stream>>>(x, posw, xs);
  // k / v projections on sampled features -> f16 directly (f32 buffers were dead)
  k_conv1x1<<<gconv, blk, 0, stream>>>(xs, kW, kb, nullptr, kh, CC, CC, NSMP, 1.0f, 1);
  k_conv1x1<<<gconv, blk, 0, stream>>>(xs, vW, vb, nullptr, vt, CC, CC, NSMP, 1.0f, 2);
  // lepe
  k_dwconv3<<<(unsigned)((NE + 255) / 256), blk, 0, stream>>>(q, rpw, rpb, lepe, BB, CC);
  // WMMA attention (softmax fused, + lepe), 2 query tiles per wave
  k_attn<<<dim3(HWP / 32, BHD), dim3(32), 0, stream>>>(qh, kh, vt, lepe, outb);
  // output projection -> y
  k_conv1x1<<<gconv, blk, 0, stream>>>(outb, oW, ob, out, nullptr, CC, CC, HWP, 1.0f, 0);
}
